// CausalSelfAttention_61787399520469
// MI455X (gfx1250) — matrix operand order repaired and checked by reading, not compiled, not run on hardware
//
#include <hip/hip_runtime.h>
#include <hip/hip_bf16.h>
#include <math.h>

// CDNA5 / gfx1250: wave32, WMMA 16x16x32 bf16 -> f32 accumulate.
typedef __attribute__((ext_vector_type(16))) __bf16 v16bf;
typedef __attribute__((ext_vector_type(8)))  float  v8f;
typedef __attribute__((ext_vector_type(4)))  int    v4i;

#define NUM_HEAD 8
#define HEAD_DIM 64

#define AS1 __attribute__((address_space(1)))
#define AS3 __attribute__((address_space(3)))

#ifndef __has_builtin
#define __has_builtin(x) 0
#endif
#if defined(__AMDGCN__) && \
    __has_builtin(__builtin_amdgcn_global_load_async_to_lds_b128) && \
    __has_builtin(__builtin_amdgcn_s_wait_asynccnt)
#define USE_ASYNC_LDS 1
#else
#define USE_ASYNC_LDS 0
#endif

// ---------------------------------------------------------------------------
// Fragment loaders for CDNA5 WMMA (wave32): A and B share one contiguous k order, not the ISA table order.
// All loaders read 16 contiguous (or 2x8-contiguous) bf16 per lane so the
// backend merges them into ds_load_b128 / global_load_b128 (no scalar u16
// loads, no v_mov_b16 packing).
// A (16x32, MxK): lane m = lane&15, half = lane>>4;
//   elements 0..15 = K (half*16 + 0..15)      -> 32 contiguous bytes
// B (32x16, KxN) from N-MAJOR storage (rows = N, cols = K):
//   lane n = lane&15, half = lane>>4; elements 0..15 = K (half*16 + 0..15)
//   -> 32 contiguous bytes per lane.
// C/D (16x16 f32): VGPR r holds M = r + 8*(lane>>4), N = lane&15.
// ---------------------------------------------------------------------------

__device__ __forceinline__ v16bf load_a_frag(const __bf16* tile, int ld, int lane) {
  const int half = lane >> 4;
  const int m    = lane & 15;
  const __bf16* row = tile + (size_t)m * ld;
  v16bf a;
#pragma unroll
  for (int j = 0; j < 16; ++j) a[j] = row[half * 16 + j];
  return a;
}

// B fragment from N-major storage (rows = N, cols = K), row stride ld.
__device__ __forceinline__ v16bf load_b_frag_t(const __bf16* tile, int ld, int lane) {
  const int half = lane >> 4;
  const int n    = lane & 15;
  const __bf16* row = tile + (size_t)n * ld + half * 16;
  v16bf b;
#pragma unroll
  for (int j = 0; j < 16; ++j) b[j] = row[j];
  return b;
}

// ---------------------------------------------------------------------------
// Stage a 64x64 bf16 tile into LDS with a 72-element (144 B, 16 B-aligned)
// row pitch. Source rows are `srcld` elements apart (64 = contiguous tile,
// T = transposed-V tile). 512 x 16 B chunks over 128 threads.
// Async path: GLOBAL_LOAD_ASYNC_TO_LDS_B128 (ASYNCcnt), no VGPR round-trip.
// Builtin signature (probed via hipcc diagnostic): (v4i as1*, v4i as3*, imm, imm).
// ---------------------------------------------------------------------------
__device__ __forceinline__ void stage_tile64(const __bf16* __restrict__ g, int srcld,
                                             __bf16* __restrict__ lds, int tid) {
#pragma unroll
  for (int c = tid; c < 512; c += 128) {
    const int row = c >> 3;
    const int col = (c & 7) * 8;
    const __bf16* src = g + (size_t)row * srcld + col;
    __bf16* dst = lds + row * 72 + col;
#if USE_ASYNC_LDS
    __builtin_amdgcn_global_load_async_to_lds_b128(
        (AS1 v4i*)(void*)src, (AS3 v4i*)(void*)dst, 0, 0);
#else
    *(uint4*)dst = *(const uint4*)src;
#endif
  }
}

__device__ __forceinline__ void wait_stage() {
#if USE_ASYNC_LDS
  __builtin_amdgcn_s_wait_asynccnt(0);
#endif
}

// ---------------------------------------------------------------------------
// Fused QKV projection: q,k,v = x @ W{q,k,v} + b{q,k,v}.
// q,k written bf16 head-split [B, H, T, 64]; v written TRANSPOSED bf16
// [B, H, 64, T] so the attention kernel's V fragments are N-major.
// Weight tiles staged N-major (transposed) in LDS -> contiguous B-frag loads.
// x tile read + converted once per K-step, reused for all 3 weights
// (12 WMMAs per K-step per wave). q pre-scaled by 1/sqrt(D).
// ---------------------------------------------------------------------------
__global__ __launch_bounds__(128) void qkv_proj_wmma(
    const float* __restrict__ x,
    const float* __restrict__ Wq, const float* __restrict__ bq,
    const float* __restrict__ Wk, const float* __restrict__ bk,
    const float* __restrict__ Wv, const float* __restrict__ bv,
    __bf16* __restrict__ qo, __bf16* __restrict__ ko, __bf16* __restrict__ vo,
    int BT, int T, int C) {
  __shared__ __bf16 At[64][40];       // 64x32 x tile (bf16), M-major, padded
  __shared__ __bf16 Wt[3][64][40];    // weight tiles, N-MAJOR [n][k], padded

  const int tid  = threadIdx.x;
  const int wave = tid >> 5;
  const int lane = tid & 31;
  const int m0   = blockIdx.x * 64;
  const int n0   = blockIdx.y * 64;   // one head's 64 columns

  const float* Ws[3] = {Wq, Wk, Wv};
  const float* bs[3] = {bq, bk, bv};

  const v8f zero = {0.f, 0.f, 0.f, 0.f, 0.f, 0.f, 0.f, 0.f};
  v8f acc[3][4];
#pragma unroll
  for (int m = 0; m < 3; ++m)
#pragma unroll
    for (int nt = 0; nt < 4; ++nt) acc[m][nt] = zero;

  for (int k0 = 0; k0 < C; k0 += 32) {
    // Stage x tile once (fp32 -> bf16), M-major.
    for (int i = tid; i < 64 * 32; i += 128) {
      const int r = i >> 5, c = i & 31;
      At[r][c] = (__bf16)x[(size_t)(m0 + r) * C + (k0 + c)];
    }
    // Stage the three weight tiles transposed: coalesced global read along n,
    // N-major LDS write.
#pragma unroll
    for (int m = 0; m < 3; ++m) {
      const float* W = Ws[m];
      for (int i = tid; i < 32 * 64; i += 128) {
        const int r = i >> 6, c = i & 63;   // r = k, c = n
        Wt[m][c][r] = (__bf16)W[(size_t)(k0 + r) * C + (n0 + c)];
      }
    }
    __syncthreads();

    const v16bf a = load_a_frag(&At[wave * 16][0], 40, lane);
#pragma unroll
    for (int m = 0; m < 3; ++m) {
#pragma unroll
      for (int nt = 0; nt < 4; ++nt) {
        const v16bf b = load_b_frag_t(&Wt[m][nt * 16][0], 40, lane);
        acc[m][nt] = __builtin_amdgcn_wmma_f32_16x16x32_bf16(
            false, a, false, b, (short)0, acc[m][nt], false, false);
      }
    }
    __syncthreads();
  }

  const int half = lane >> 4;
  const int nlo  = lane & 15;
  const int h    = n0 / HEAD_DIM;  // tile width == head dim
#pragma unroll
  for (int m = 0; m < 3; ++m) {
    const float scale = (m == 0) ? 0.125f : 1.0f;  // fold 1/sqrt(D) into q
#pragma unroll
    for (int nt = 0; nt < 4; ++nt) {
      const int col = n0 + nt * 16 + nlo;
      const int d   = col - n0;
      const float bval = bs[m][col];
#pragma unroll
      for (int r = 0; r < 8; ++r) {
        const int row = m0 + wave * 16 + r + 8 * half;
        const int bb = row / T, t = row % T;
        const float val = (acc[m][nt][r] + bval) * scale;
        const size_t headbase = ((size_t)bb * NUM_HEAD + h);
        if (m == 0)      qo[(headbase * T + t) * HEAD_DIM + d] = (__bf16)val;
        else if (m == 1) ko[(headbase * T + t) * HEAD_DIM + d] = (__bf16)val;
        else             vo[(headbase * HEAD_DIM + d) * T + t] = (__bf16)val;  // transposed
      }
    }
  }
}

// ---------------------------------------------------------------------------
// Output projection: out[BT x C] (fp32) = y[BT x C] (bf16) @ Wp + bp.
// Weight tile staged N-major for contiguous B-fragment loads.
// ---------------------------------------------------------------------------
__global__ __launch_bounds__(128) void out_proj_wmma(
    const __bf16* __restrict__ A, const float* __restrict__ W,
    const float* __restrict__ bias, float* __restrict__ O,
    int BT, int C) {
  __shared__ __bf16 At[64][40];
  __shared__ __bf16 Wt[64][40];  // N-MAJOR [n][k]

  const int tid  = threadIdx.x;
  const int wave = tid >> 5;
  const int lane = tid & 31;
  const int m0   = blockIdx.x * 64;
  const int n0   = blockIdx.y * 64;

  const v8f zero = {0.f, 0.f, 0.f, 0.f, 0.f, 0.f, 0.f, 0.f};
  v8f acc[4] = {zero, zero, zero, zero};

  for (int k0 = 0; k0 < C; k0 += 32) {
    for (int i = tid; i < 64 * 32; i += 128) {
      const int r = i >> 5, c = i & 31;
      At[r][c] = A[(size_t)(m0 + r) * C + (k0 + c)];
    }
    for (int i = tid; i < 32 * 64; i += 128) {
      const int r = i >> 6, c = i & 63;   // r = k, c = n
      Wt[c][r] = (__bf16)W[(size_t)(k0 + r) * C + (n0 + c)];
    }
    __syncthreads();

    const v16bf a = load_a_frag(&At[wave * 16][0], 40, lane);
#pragma unroll
    for (int nt = 0; nt < 4; ++nt) {
      const v16bf b = load_b_frag_t(&Wt[nt * 16][0], 40, lane);
      acc[nt] = __builtin_amdgcn_wmma_f32_16x16x32_bf16(
          false, a, false, b, (short)0, acc[nt], false, false);
    }
    __syncthreads();
  }

  const int half = lane >> 4;
  const int nlo  = lane & 15;
#pragma unroll
  for (int nt = 0; nt < 4; ++nt) {
    const int col = n0 + nt * 16 + nlo;
    const float bval = bias[col];
#pragma unroll
    for (int r = 0; r < 8; ++r) {
      const int row = m0 + wave * 16 + r + 8 * half;
      O[(size_t)row * C + col] = acc[nt][r] + bval;
    }
  }
}

// ---------------------------------------------------------------------------
// Flash attention (no causal mask, matching reference). Per block: one (b,h),
// one 64-row Q tile; stream over 64-key tiles with online softmax. K and
// transposed-V tiles staged into LDS once per block (double-buffered, async
// global->LDS). K tile is [key][d] (N-major for S), V tile is [d][key]
// (N-major for P@V) -> all fragment reads are contiguous ds_load_b128.
// q arrives pre-scaled by 1/sqrt(D).
// ---------------------------------------------------------------------------
__global__ __launch_bounds__(128) void attn_flash_wmma(
    const __bf16* __restrict__ q, const __bf16* __restrict__ k,
    const __bf16* __restrict__ vT, __bf16* __restrict__ y,
    int B, int T) {
  __shared__ __align__(16) __bf16 Kt[2][64][72];  // [key][d], double-buffered
  __shared__ __align__(16) __bf16 Vt[2][64][72];  // [d][key], double-buffered
  __shared__ __align__(16) __bf16 Pt[64][72];     // P tile: C-layout -> A-layout bridge

  const int C    = NUM_HEAD * HEAD_DIM;
  const int tid  = threadIdx.x;
  const int wave = tid >> 5;
  const int lane = tid & 31;
  const int half = lane >> 4;
  const int nlo  = lane & 15;

  const int bh = blockIdx.y;
  const int b  = bh / NUM_HEAD;
  const int h  = bh % NUM_HEAD;
  const int m0 = blockIdx.x * 64;

  const __bf16* qb = q  + (size_t)bh * T * HEAD_DIM;
  const __bf16* kb = k  + (size_t)bh * T * HEAD_DIM;
  const __bf16* vb = vT + (size_t)bh * HEAD_DIM * T;  // [d][t]

  // Q fragments for this wave's 16 rows (K = 64 -> two 32-deep fragments).
  const v16bf qa0 = load_a_frag(qb + (size_t)(m0 + wave * 16) * HEAD_DIM + 0,  HEAD_DIM, lane);
  const v16bf qa1 = load_a_frag(qb + (size_t)(m0 + wave * 16) * HEAD_DIM + 32, HEAD_DIM, lane);

  const v8f zero = {0.f, 0.f, 0.f, 0.f, 0.f, 0.f, 0.f, 0.f};
  v8f oacc[4] = {zero, zero, zero, zero};
  float rm[8], rl[8];
#pragma unroll
  for (int r = 0; r < 8; ++r) { rm[r] = -1e30f; rl[r] = 0.f; }

  const int KT = T / 64;

  // Prologue: stage tile 0 into buffer 0.
  stage_tile64(kb, HEAD_DIM, &Kt[0][0][0], tid);   // rows = keys (contiguous)
  stage_tile64(vb, T,        &Vt[0][0][0], tid);   // rows = d, stride T

  for (int kt = 0; kt < KT; ++kt) {
    const int cur = kt & 1;
    wait_stage();        // s_wait_asynccnt 0: this wave's staged chunks landed
    __syncthreads();     // all waves' chunks landed

    if (kt + 1 < KT) {   // issue next tile into the other buffer (prefetch depth 1)
      stage_tile64(kb + (size_t)(kt + 1) * 64 * HEAD_DIM, HEAD_DIM, &Kt[cur ^ 1][0][0], tid);
      stage_tile64(vb + (size_t)(kt + 1) * 64,            T,        &Vt[cur ^ 1][0][0], tid);
    }

    // S = Q @ K^T  (64 keys: 4 N-tiles x 2 K-steps), K from LDS.
    v8f s[4] = {zero, zero, zero, zero};
#pragma unroll
    for (int nt = 0; nt < 4; ++nt) {
      const v16bf b0 = load_b_frag_t(&Kt[cur][nt * 16][0], 72, lane);
      s[nt] = __builtin_amdgcn_wmma_f32_16x16x32_bf16(
          false, qa0, false, b0, (short)0, s[nt], false, false);
      const v16bf b1 = load_b_frag_t(&Kt[cur][nt * 16][32], 72, lane);
      s[nt] = __builtin_amdgcn_wmma_f32_16x16x32_bf16(
          false, qa1, false, b1, (short)0, s[nt], false, false);
    }

    // Online softmax. C-layout row = r + 8*half lives on the 16 lanes sharing
    // `half`, so xor-shuffles with masks 1,2,4,8 reduce exactly one row.
#pragma unroll
    for (int r = 0; r < 8; ++r) {
      float mx = -1e30f;
#pragma unroll
      for (int nt = 0; nt < 4; ++nt) mx = fmaxf(mx, s[nt][r]);
#pragma unroll
      for (int o = 1; o < 16; o <<= 1) mx = fmaxf(mx, __shfl_xor(mx, o, 32));
      const float mnew = fmaxf(rm[r], mx);
      const float corr = __expf(rm[r] - mnew);
      rm[r] = mnew;

      const int prow = wave * 16 + r + 8 * half;
      float rs = 0.f;
#pragma unroll
      for (int nt = 0; nt < 4; ++nt) {
        const float p = __expf(s[nt][r] - mnew);
        rs += p;
        Pt[prow][nt * 16 + nlo] = (__bf16)p;
        oacc[nt][r] *= corr;
      }
#pragma unroll
      for (int o = 1; o < 16; o <<= 1) rs += __shfl_xor(rs, o, 32);
      rl[r] = rl[r] * corr + rs;
    }
    __syncthreads();

    // O += P @ V  (contraction over 64 keys: 2 K-steps x 4 N-tiles), V from LDS.
#pragma unroll
    for (int ks = 0; ks < 2; ++ks) {
      const v16bf pa = load_a_frag(&Pt[wave * 16][ks * 32], 72, lane);
#pragma unroll
      for (int nt = 0; nt < 4; ++nt) {
        const v16bf vfrag = load_b_frag_t(&Vt[cur][nt * 16][ks * 32], 72, lane);
        oacc[nt] = __builtin_amdgcn_wmma_f32_16x16x32_bf16(
            false, pa, false, vfrag, (short)0, oacc[nt], false, false);
      }
    }
    // Next iteration's top-of-loop barrier orders Pt/K/V buffer reuse.
  }

  // Normalize and write back to merged-head layout [B, T, C] (bf16 for Wp GEMM).
#pragma unroll
  for (int nt = 0; nt < 4; ++nt) {
#pragma unroll
    for (int r = 0; r < 8; ++r) {
      const int t = m0 + wave * 16 + r + 8 * half;
      const int c = h * HEAD_DIM + nt * 16 + nlo;
      const float val = oacc[nt][r] / rl[r];
      y[((size_t)b * T + t) * C + c] = (__bf16)val;
    }
  }
}

// ---------------------------------------------------------------------------
// Launch: fused QKV projection -> flash attention -> output projection.
// Workspace: q,k (head-split), vT (head-split transposed), y: 4 x 8 MB bf16.
// ---------------------------------------------------------------------------
extern "C" void kernel_launch(void* const* d_in, const int* in_sizes, int n_in,
                              void* d_out, int out_size, void* d_ws, size_t ws_size,
                              hipStream_t stream) {
  const float* x  = (const float*)d_in[0];
  const float* Wq = (const float*)d_in[1];
  const float* bq = (const float*)d_in[2];
  const float* Wk = (const float*)d_in[3];
  const float* bk = (const float*)d_in[4];
  const float* Wv = (const float*)d_in[5];
  const float* bv = (const float*)d_in[6];
  const float* Wp = (const float*)d_in[7];
  const float* bp = (const float*)d_in[8];

  const int C  = 512;                 // matches in_sizes[1] = C*C
  const int T  = 2048;
  const int BT = in_sizes[0] / C;     // 8192
  const int B  = BT / T;              // 4

  __bf16* qd = (__bf16*)d_ws;
  __bf16* kd = qd + (size_t)BT * C;
  __bf16* vd = kd + (size_t)BT * C;   // transposed [B,H,D,T]
  __bf16* yd = vd + (size_t)BT * C;

  const dim3 blk(128);
  const dim3 gproj(BT / 64, C / 64);

  qkv_proj_wmma<<<gproj, blk, 0, stream>>>(x, Wq, bq, Wk, bk, Wv, bv,
                                           qd, kd, vd, BT, T, C);

  const dim3 gattn(T / 64, B * NUM_HEAD);
  attn_flash_wmma<<<gattn, blk, 0, stream>>>(qd, kd, vd, yd, B, T);

  out_proj_wmma<<<gproj, blk, 0, stream>>>(yd, Wp, bp, (float*)d_out, BT, C);
}
